// CNN_30210799960376
// MI455X (gfx1250) — matrix operand order repaired and checked by reading, not compiled, not run on hardware
//
#include <hip/hip_runtime.h>

// ---------------------------------------------------------------------------
// MI455X (gfx1250): implicit-GEMM 3D conv via v_wmma_f32_16x16x32_f16.
//  - K row-padded (7 -> 8): every 8-run of K = one contiguous input row.
//  - Activations halo-padded => no bounds checks, no divergence.
//  - NT=4 N-tiles per wave: one A-fragment feeds 4 WMMAs per K-tile.
//  - Row offsets precomputed in LDS: inner loop = 2 ds_load broadcasts,
//    8 address adds, 9 vector loads, 4 WMMAs.
// ---------------------------------------------------------------------------

typedef __attribute__((ext_vector_type(16))) _Float16 v16h;
typedef __attribute__((ext_vector_type(8)))  _Float16 v8h;
typedef __attribute__((ext_vector_type(8)))  float    v8f;

#define KS   7
#define KV   343
#define NB   16
#define EPSN 1e-5f
#define NT   4             // N-tiles per wave (register blocking)
#define WPB  8             // waves per 256-thread block
#define MAXROWS 4096       // >= max Cin*49 (block2 conv2: 80*49 = 3920)

// --------------------------------------------------------------------------
__global__ __launch_bounds__(256) void k_fill_zero_f16(
    _Float16* __restrict__ p, long long n)
{
    long long i = (long long)blockIdx.x * 256 + threadIdx.x;
    if (i < n) p[i] = (_Float16)0.f;
}

// --------------------------------------------------------------------------
// Per-channel mean / inv-std over (n, d, h, w). One workgroup per channel.
// --------------------------------------------------------------------------
__global__ __launch_bounds__(256) void k_channel_stats(
    const float* __restrict__ y, float* __restrict__ stats,
    int N, int C, int D3)
{
    int c = blockIdx.x;
    float s = 0.f, s2 = 0.f;
    for (int n = 0; n < N; ++n) {
        const float* p = y + ((long long)n * C + c) * D3;
        for (int i = threadIdx.x; i < D3; i += 256) {
            float v = p[i];
            s += v; s2 += v * v;
        }
    }
    __shared__ float ls[256], lq[256];
    ls[threadIdx.x] = s; lq[threadIdx.x] = s2;
    __syncthreads();
    for (int off = 128; off > 0; off >>= 1) {
        if (threadIdx.x < off) {
            ls[threadIdx.x] += ls[threadIdx.x + off];
            lq[threadIdx.x] += lq[threadIdx.x + off];
        }
        __syncthreads();
    }
    if (threadIdx.x == 0) {
        float cnt = (float)N * (float)D3;
        float mu  = ls[0] / cnt;
        float var = lq[0] / cnt - mu * mu;
        stats[2 * c]     = mu;
        stats[2 * c + 1] = rsqrtf(var + EPSN);
    }
}

// --------------------------------------------------------------------------
// Normalize (+ optional bias/ReLU) and scatter into the halo-padded FP16
// activation tensor [N][C][Dp][Dp][Dp] (interior at +pad per dim).
// --------------------------------------------------------------------------
__global__ __launch_bounds__(256) void k_norm_act_pad(
    const float* __restrict__ y, const float* __restrict__ stats,
    const float* __restrict__ b2, int useRelu,
    _Float16* __restrict__ out, int N, int C, int D, int Dp, int pad)
{
    long long total = (long long)N * C * D * D * D;
    long long i = (long long)blockIdx.x * 256 + threadIdx.x;
    if (i >= total) return;
    int x = (int)(i % D); long long t = i / D;
    int yy = (int)(t % D); t /= D;
    int z = (int)(t % D); t /= D;
    int c = (int)(t % C);
    int nc = (int)t;   // n*C + c
    float v = (y[i] - stats[2 * c]) * stats[2 * c + 1];
    if (useRelu) v = fmaxf(v + b2[c], 0.f);
    long long o = (((long long)nc * Dp + (z + pad)) * Dp + (yy + pad)) * Dp + (x + pad);
    out[o] = (_Float16)v;
}

// --------------------------------------------------------------------------
// WMMA 16-bit A-fragment element -> K offset inside a 32-wide K tile.
// --------------------------------------------------------------------------
__device__ __forceinline__ int wmma_k_in_tile_A(int e, int hi)
{
    return hi * 16 + e;
}

// --------------------------------------------------------------------------
// Conv1 weights -> packed A fragments, row-padded K (row=cin*49+kd*7+kh, kw<8;
// kw==7 and padded rows get zero).  wpack[((mt*Kt + kt)*32 + lane)*16 + e]
// --------------------------------------------------------------------------
__global__ __launch_bounds__(256) void k_prep_w1(
    const float* __restrict__ w1_ss, const float* __restrict__ w1_vs,
    const float* __restrict__ B11a,  const float* __restrict__ B31,
    _Float16* __restrict__ wpack,
    int s_out, int vec, int s_in, int Mtiles, int Ktiles)
{
    long long idx   = (long long)blockIdx.x * 256 + threadIdx.x;
    long long total = (long long)Mtiles * Ktiles * 512;
    if (idx >= total) return;
    int e    = (int)(idx & 15);
    int lane = (int)((idx >> 4) & 31);
    long long t = idx >> 9;
    int kt    = (int)(t % Ktiles);
    int mtile = (int)(t / Ktiles);
    int m  = mtile * 16 + (lane & 15);
    int hi = lane >> 4;
    int K  = kt * 32 + wmma_k_in_tile_A(e, hi);
    int row = K >> 3, kw = K & 7;
    int Cout = s_out + 3 * vec;
    float val = 0.f;
    if (m < Cout && kw < 7 && row < s_in * 49) {
        int q = row / 49, rr = row % 49;
        int kd = rr / 7, kh = rr % 7;
        int dhw = kd * 49 + kh * 7 + kw;
        if (m < s_out) {
            #pragma unroll
            for (int b = 0; b < 4; ++b)
                val += w1_ss[(m * s_in + q) * 4 + b] * B11a[b * KV + dhw];
        } else {
            int pv = m - s_out, vq = pv / 3, vi = pv % 3;
            #pragma unroll
            for (int b = 0; b < 4; ++b)
                val += w1_vs[(vq * s_in + q) * 4 + b] * B31[(b * 3 + vi) * KV + dhw];
        }
    }
    wpack[idx] = (_Float16)val;
}

// --------------------------------------------------------------------------
// Conv2 weights (ss | sv | st contractions) -> packed A fragments.
// --------------------------------------------------------------------------
__global__ __launch_bounds__(256) void k_prep_w2(
    const float* __restrict__ w2_ss, const float* __restrict__ w2_sv,
    const float* __restrict__ w2_st,
    const float* __restrict__ B11b, const float* __restrict__ B13,
    const float* __restrict__ B19,
    _Float16* __restrict__ wpack,
    int s_out, int vec, int Mtiles, int Ktiles)
{
    long long idx   = (long long)blockIdx.x * 256 + threadIdx.x;
    long long total = (long long)Mtiles * Ktiles * 512;
    if (idx >= total) return;
    int e    = (int)(idx & 15);
    int lane = (int)((idx >> 4) & 31);
    long long t = idx >> 9;
    int kt    = (int)(t % Ktiles);
    int mtile = (int)(t / Ktiles);
    int m  = mtile * 16 + (lane & 15);
    int hi = lane >> 4;
    int K  = kt * 32 + wmma_k_in_tile_A(e, hi);
    int row = K >> 3, kw = K & 7;
    int Csvt = s_out + 12 * vec;
    float val = 0.f;
    if (m < s_out && kw < 7 && row < Csvt * 49) {
        int cin = row / 49, rr = row % 49;
        int kd = rr / 7, kh = rr % 7;
        int dhw = kd * 49 + kh * 7 + kw;
        if (cin < s_out) {
            #pragma unroll
            for (int b = 0; b < 4; ++b)
                val += w2_ss[(m * s_out + cin) * 4 + b] * B11b[b * KV + dhw];
        } else if (cin < s_out + 3 * vec) {
            int cv = cin - s_out, q = cv / 3, j = cv % 3;
            #pragma unroll
            for (int b = 0; b < 4; ++b)
                val += w2_sv[(m * vec + q) * 4 + b] * B13[(b * 3 + j) * KV + dhw];
        } else {
            int ct = cin - s_out - 3 * vec, q = ct / 9, j = ct % 9;
            #pragma unroll
            for (int b = 0; b < 12; ++b)
                val += w2_st[(m * vec + q) * 12 + b] * B19[(b * 9 + j) * KV + dhw];
        }
    }
    wpack[idx] = (_Float16)val;
}

// --------------------------------------------------------------------------
// Implicit-GEMM conv on halo-padded input [N][Cin][Dp][Dp][Dp].
//   B layout (contiguous K order, same as the A packing):
//   element e of lane (n=lane&15, hi=lane>>4) holds K = kt*32 + hi*16 + e,
//   i.e. rows (kt*4 + 2*hi) and (kt*4 + 2*hi + 1), kw = e&7 contiguous.
//   Row offsets (cin*Dp3 + kd*Dp2 + kh*Dp) served from an LDS table.
// --------------------------------------------------------------------------
__global__ __launch_bounds__(256) void k_conv_wmma(
    const _Float16* __restrict__ in, const _Float16* __restrict__ wpack,
    float* __restrict__ out,
    int N, int Cin, int Dp, int Cout, int Dout, int stride,
    int Ktiles, int rowsTotal /* = Cin*49 */)
{
    const int lane   = threadIdx.x & 31;
    const int wave   = threadIdx.x >> 5;
    const int mtile  = blockIdx.y;
    const int ntile0 = (blockIdx.x * WPB + wave) * NT;

    const int D2 = Dout * Dout, D3 = D2 * Dout;
    const int totalPos = N * D3;
    const int hi  = lane >> 4;
    const int Dp2 = Dp * Dp;
    const int Dp3 = Dp2 * Dp;

    // LDS row-offset table: rowOff(row) for row in [0, rowsTotal)
    __shared__ int rofftab[MAXROWS];
    for (int r = threadIdx.x; r < rowsTotal; r += 256) {
        int cin = r / 49, rr = r - cin * 49;
        int kd = rr / 7,  kh = rr - kd * 7;
        rofftab[r] = cin * Dp3 + kd * Dp2 + kh * Dp;
    }
    __syncthreads();

    int  base[NT], obase[NT];
    bool valid[NT];
    #pragma unroll
    for (int c = 0; c < NT; ++c) {
        int col = (ntile0 + c) * 16 + (lane & 15);
        valid[c] = col < totalPos;
        if (col > totalPos - 1) col = totalPos - 1;   // clamp tail to a safe address
        int nb = col / D3; int r = col - nb * D3;
        int od = r / D2;   r -= od * D2;
        int oh = r / Dout; int ow = r - oh * Dout;
        base[c]  = nb * Cin * Dp3 + (od * stride) * Dp2 + (oh * stride) * Dp + ow * stride;
        obase[c] = nb * Cout * D3 + (od * D2 + oh * Dout + ow);
    }

    const v16h* __restrict__ wv =
        (const v16h*)wpack + (long long)mtile * Ktiles * 32;

    union BU { v16h v; v8h h[2]; };

    v8f acc[NT] = {};
    #pragma unroll 2
    for (int kt = 0; kt < Ktiles; ++kt) {
        if (kt + 4 < Ktiles)
            __builtin_prefetch((const void*)(wv + (long long)(kt + 4) * 32 + lane), 0, 1);
        v16h a = wv[(long long)kt * 32 + lane];   // one 32B A-fragment load

        BU bf[NT];
        const int row0 = kt * 4 + hi * 2;
        #pragma unroll
        for (int run = 0; run < 2; ++run) {
            int row = row0 + run;
            if (row > rowsTotal - 1) row = rowsTotal - 1;  // padded rows: weights are 0
            const int roff = rofftab[row];                  // LDS broadcast read
            #pragma unroll
            for (int c = 0; c < NT; ++c) {
                __builtin_memcpy(&bf[c].h[run], in + base[c] + roff, 16);
            }
        }

        #pragma unroll
        for (int c = 0; c < NT; ++c) {
            acc[c] = __builtin_amdgcn_wmma_f32_16x16x32_f16(
                /*neg_a=*/false, a, /*neg_b=*/false, bf[c].v,
                /*c_mod=*/(short)0, acc[c], /*reuse_a=*/false, /*reuse_b=*/false);
        }
    }

    #pragma unroll
    for (int c = 0; c < NT; ++c) {
        if (valid[c]) {
            #pragma unroll
            for (int r = 0; r < 8; ++r) {
                int m = mtile * 16 + r + 8 * hi;   // C/D layout: hi half holds M+8
                if (m < Cout) out[obase[c] + m * D3] = acc[c][r];
            }
        }
    }
}

// --------------------------------------------------------------------------
// Assemble svt = [relu(s + b1) | v | outer(v3,v3)] into the halo-padded
// FP16 tensor for conv2 (pad = 3).
// --------------------------------------------------------------------------
__global__ __launch_bounds__(256) void k_assemble_svt_pad(
    const float* __restrict__ sv, const float* __restrict__ b1,
    _Float16* __restrict__ svt, int N, int s_out, int vec, int D, int Dp)
{
    const int pad = 3;
    int Csv  = s_out + 3 * vec;
    int Csvt = s_out + 12 * vec;
    int D3 = D * D * D;
    long long total = (long long)N * Csvt * D3;
    long long i = (long long)blockIdx.x * 256 + threadIdx.x;
    if (i >= total) return;
    int x = (int)(i % D); long long t = i / D;
    int yy = (int)(t % D); t /= D;
    int z = (int)(t % D); t /= D;
    int c = (int)(t % Csvt);
    int n = (int)(t / Csvt);
    int sp = (z * D + yy) * D + x;
    const float* svn = sv + (long long)n * Csv * D3;
    float val;
    if (c < s_out) {
        val = fmaxf(svn[(long long)c * D3 + sp] + b1[c], 0.f);
    } else if (c < s_out + 3 * vec) {
        val = svn[(long long)c * D3 + sp];
    } else {
        int ct = c - s_out - 3 * vec;
        int q = ct / 9, ij = ct % 9, ii = ij / 3, jj = ij % 3;
        float vi = svn[(long long)(s_out + q * 3 + ii) * D3 + sp];
        float vj = svn[(long long)(s_out + q * 3 + jj) * D3 + sp];
        val = vi * vj;
    }
    long long o = (((long long)(n * Csvt + c) * Dp + (z + pad)) * Dp + (yy + pad)) * Dp + (x + pad);
    svt[o] = (_Float16)val;
}

// --------------------------------------------------------------------------
// Final head: spatial mean of normalized y, batch-dim norm, gamma/beta.
// --------------------------------------------------------------------------
__global__ __launch_bounds__(256) void k_final(
    const float* __restrict__ y, const float* __restrict__ stats,
    const float* __restrict__ gamma, const float* __restrict__ beta,
    float* __restrict__ out, int N, int C, int D3)
{
    __shared__ float pm[256];
    int t = threadIdx.x;
    if (t < N * C) {
        int n = t / C, c = t % C;
        const float* p = y + ((long long)n * C + c) * D3;
        float s = 0.f;
        for (int i = 0; i < D3; ++i) s += p[i];
        pm[t] = (s / (float)D3 - stats[2 * c]) * stats[2 * c + 1];
    }
    __syncthreads();
    if (t < N * C) {
        int c = t % C;
        float mu = 0.f;
        for (int nn = 0; nn < N; ++nn) mu += pm[nn * C + c];
        mu /= (float)N;
        float var = 0.f;
        for (int nn = 0; nn < N; ++nn) { float d = pm[nn * C + c] - mu; var += d * d; }
        var /= (float)N;
        out[t] = gamma[c] * (pm[t] - mu) * rsqrtf(var + EPSN) + beta[c];
    }
}

// ---------------------------------------------------------------------------
// Host orchestration
// ---------------------------------------------------------------------------
extern "C" void kernel_launch(void* const* d_in, const int* in_sizes, int n_in,
                              void* d_out, int out_size, void* d_ws, size_t ws_size,
                              hipStream_t stream)
{
    (void)in_sizes; (void)n_in; (void)out_size; (void)ws_size;

    // pytree (sorted-key) flatten: d_in[0]=x, [1]=beta, blocks @ 2+12b ->
    // [B11a,B11b,B13,B19,B31,b1,b2,w1_ss,w1_vs,w2_ss,w2_st,w2_sv], [50]=gamma
    const float* X     = (const float*)d_in[0];
    const float* beta  = (const float*)d_in[1];
    const float* gamma = (const float*)d_in[50];

    char* ws = (char*)d_ws;
    size_t off = 0;
    auto carve = [&](size_t bytes) { size_t o = off; off += (bytes + 255) & ~(size_t)255; return o; };
    _Float16* act  = (_Float16*)(ws + carve((size_t)6483584  * 2));  // padded conv1 input (max: b0 16*74^3)
    _Float16* svt  = (_Float16*)(ws + carve((size_t)28672000 * 2));  // padded conv2 input (max: b0 448*40^3)
    float*    sv   = (float*)   (ws + carve((size_t)6288640  * 4));  // conv1 out (max: b0 16*10*34^3)
    float*    yb   = (float*)   (ws + carve((size_t)2515456  * 4));  // conv2 out (max: b0 16*4*34^3)
    _Float16* w1p  = (_Float16*)(ws + carve((size_t)401408   * 2));  // packed W1 (max: b3 2*392*512)
    _Float16* w2p  = (_Float16*)(ws + carve((size_t)1003520  * 2));  // packed W2 (max: b2 2*980*512)
    float*    stats= (float*)   (ws + carve(256 * sizeof(float)));

    static const int SO[4] = {4, 16, 32, 10};
    static const int VE[4] = {2, 4, 4, 4};
    static const int SI[4] = {1, 4, 16, 32};

    // ---- input normalization -> padded FP16 ----
    int Din = 64;
    {
        const int D3 = Din * Din * Din;
        const int Dp = Din + 10;
        k_channel_stats<<<1, 256, 0, stream>>>(X, stats, NB, 1, D3);
        long long padTot = (long long)NB * 1 * Dp * Dp * Dp;
        k_fill_zero_f16<<<(unsigned)((padTot + 255) / 256), 256, 0, stream>>>(act, padTot);
        long long tot = (long long)NB * D3;
        k_norm_act_pad<<<(unsigned)((tot + 255) / 256), 256, 0, stream>>>(
            X, stats, nullptr, 0, act, NB, 1, Din, Dp, 5);
    }

    for (int b = 0; b < 4; ++b) {
        const int base = 2 + 12 * b;
        const float* B11a  = (const float*)d_in[base + 0];
        const float* B11b  = (const float*)d_in[base + 1];
        const float* B13   = (const float*)d_in[base + 2];
        const float* B19   = (const float*)d_in[base + 3];
        const float* B31   = (const float*)d_in[base + 4];
        const float* b1    = (const float*)d_in[base + 5];
        const float* b2    = (const float*)d_in[base + 6];
        const float* w1_ss = (const float*)d_in[base + 7];
        const float* w1_vs = (const float*)d_in[base + 8];
        const float* w2_ss = (const float*)d_in[base + 9];
        const float* w2_st = (const float*)d_in[base + 10];
        const float* w2_sv = (const float*)d_in[base + 11];

        const int s_out = SO[b], vec = VE[b], Cin = SI[b];
        const int Dout  = (Din + 2 * 5 - KS) / 2 + 1;
        const int D3    = Dout * Dout * Dout;
        const int Csv   = s_out + 3 * vec;
        const int Csvt  = s_out + 12 * vec;
        const int Dp1   = Din + 10;     // conv1 halo (pad 5)
        const int DpS   = Dout + 6;     // conv2 halo (pad 3)

        // ---- conv1: M=Csv, K(row-padded)=Cin*392 ----
        const int Mt1 = (Csv + 15) / 16;
        const int Kt1 = (Cin * 392 + 31) / 32;
        {
            long long tw = (long long)Mt1 * Kt1 * 512;
            k_prep_w1<<<(unsigned)((tw + 255) / 256), 256, 0, stream>>>(
                w1_ss, w1_vs, B11a, B31, w1p, s_out, vec, Cin, Mt1, Kt1);
        }
        const int ntiles = (NB * D3 + 15) / 16;
        const int tilesPerBlk = WPB * NT;
        {
            dim3 g((ntiles + tilesPerBlk - 1) / tilesPerBlk, Mt1);
            k_conv_wmma<<<g, 256, 0, stream>>>(
                act, w1p, sv, NB, Cin, Dp1, Csv, Dout, 2, Kt1, Cin * 49);
        }

        // ---- svt assembly into padded conv2 input ----
        {
            long long padTot = (long long)NB * Csvt * DpS * DpS * DpS;
            k_fill_zero_f16<<<(unsigned)((padTot + 255) / 256), 256, 0, stream>>>(svt, padTot);
            long long tot = (long long)NB * Csvt * D3;
            k_assemble_svt_pad<<<(unsigned)((tot + 255) / 256), 256, 0, stream>>>(
                sv, b1, svt, NB, s_out, vec, Dout, DpS);
        }

        // ---- conv2: M=s_out, K(row-padded)=Csvt*392 ----
        const int Mt2 = (s_out + 15) / 16;
        const int Kt2 = (Csvt * 392 + 31) / 32;
        {
            long long tw = (long long)Mt2 * Kt2 * 512;
            k_prep_w2<<<(unsigned)((tw + 255) / 256), 256, 0, stream>>>(
                w2_ss, w2_sv, w2_st, B11b, B13, B19, w2p, s_out, vec, Mt2, Kt2);
        }
        {
            dim3 g((ntiles + tilesPerBlk - 1) / tilesPerBlk, Mt2);
            k_conv_wmma<<<g, 256, 0, stream>>>(
                svt, w2p, yb, NB, Csvt, DpS, s_out, Dout, 1, Kt2, Csvt * 49);
        }

        // ---- per-channel normalization ----
        k_channel_stats<<<s_out, 256, 0, stream>>>(yb, stats, NB, s_out, D3);
        if (b < 3) {
            const int DpN = Dout + 10;   // next block's conv1 halo
            long long padTot = (long long)NB * s_out * DpN * DpN * DpN;
            k_fill_zero_f16<<<(unsigned)((padTot + 255) / 256), 256, 0, stream>>>(act, padTot);
            long long tot = (long long)NB * s_out * D3;
            k_norm_act_pad<<<(unsigned)((tot + 255) / 256), 256, 0, stream>>>(
                yb, stats, b2, 1, act, NB, s_out, Dout, DpN, 5);
        } else {
            k_final<<<1, 256, 0, stream>>>(
                yb, stats, gamma, beta, (float*)d_out, NB, s_out, D3);
        }
        Din = Dout;
    }
}
